// WiKG_6012954214888
// MI455X (gfx1250) — matrix operand order repaired and checked by reading, not compiled, not run on hardware
//
#include <hip/hip_runtime.h>
#include <math.h>

// ---------------------------------------------------------------- types ----
typedef __bf16 bf16t;
typedef bf16t v16bf __attribute__((ext_vector_type(16)));
typedef float v8f   __attribute__((ext_vector_type(8)));
typedef unsigned short u16;

#define NN   8192
#define DIN  1024
#define DD   512
#define DH   256
#define KTOP 6
#define SCALEF 0.04419417382415922f   // 512^-0.5

#define WMMA_BF16(a, b, c) \
  __builtin_amdgcn_wmma_f32_16x16x32_bf16(false, (a), false, (b), (short)0, (c), false, false)

// ------------------------------------------------------------ bf16 utils ---
__device__ inline u16 f2bf(float f) {           // RNE float -> bf16 bits
  unsigned u = __float_as_uint(f);
  u += 0x7fffu + ((u >> 16) & 1u);
  return (u16)(u >> 16);
}

union U4BF { uint4 u; bf16t h[8]; };

// A fragment (16x32 bf16): lane holds row m=lane%16; 8 contiguous K at
// base=(lane<16?0:8) and at 16+base. p already includes the base offset.
__device__ inline v16bf load_frag_a(const u16* p) {
  U4BF lo, hi;
  lo.u = *reinterpret_cast<const uint4*>(p);
  hi.u = *reinterpret_cast<const uint4*>(p + 16);
  v16bf a;
#pragma unroll
  for (int i = 0; i < 8; ++i) { a[i] = lo.h[i]; a[i + 8] = hi.h[i]; }
  return a;
}
// B fragment (32x16 bf16): lane holds column n=lane%16; 8 K rows at
// base=(lane<16?0:8) and at 16+base (row stride ld). p includes base and n.
__device__ inline v16bf load_frag_b(const u16* p, int ld) {
  v16bf b;
#pragma unroll
  for (int i = 0; i < 8; ++i) {
    b[i]     = __builtin_bit_cast(bf16t, p[i * ld]);
    b[i + 8] = __builtin_bit_cast(bf16t, p[(16 + i) * ld]);
  }
  return b;
}

// CDNA5 async global->LDS copy (16B per lane), tracked by ASYNCcnt.
__device__ inline void async_g2l_b128(unsigned lds_off, unsigned gvoff, const void* gbase) {
  asm volatile("global_load_async_to_lds_b128 %0, %1, %2"
               :
               : "v"(lds_off), "v"(gvoff), "s"(gbase)
               : "memory");
}
__device__ inline void wait_async0() {
  asm volatile("s_wait_asynccnt 0" ::: "memory");
}

// Branchless descending top-K insert; constant indices keep arrays in VGPRs.
__device__ inline void topk_insert(float v, int id, float* tv, int* tix) {
  if (v > tv[KTOP - 1]) {
#pragma unroll
    for (int k = KTOP - 1; k >= 1; --k) {
      bool gt  = v > tv[k];
      bool gtp = v > tv[k - 1];
      tv[k]  = gt ? (gtp ? tv[k - 1]  : v)  : tv[k];
      tix[k] = gt ? (gtp ? tix[k - 1] : id) : tix[k];
    }
    bool g0 = v > tv[0];
    tix[0] = g0 ? id : tix[0];
    tv[0]  = g0 ? v  : tv[0];
  }
}

// ---------------------------------------------------------------- kernels --
__global__ void k_f32_to_bf16(const float* __restrict__ src, u16* __restrict__ dst, int n) {
  int i = blockIdx.x * blockDim.x + threadIdx.x;
  if (i < n) dst[i] = f2bf(src[i]);
}

// Generic bf16 WMMA GEMM: C[M,N] = act(A[M,K] @ B[K,N] + bias)
// One wave -> 16x64 output tile; fragments software-pipelined one K-step.
__global__ __launch_bounds__(256) void k_gemm_bf16(
    const u16* __restrict__ A, const u16* __restrict__ B,
    const float* __restrict__ bias, float* __restrict__ C,
    int M, int N, int K, int act)
{
  int wave = (blockIdx.x * 256 + (int)threadIdx.x) >> 5;
  int lane = threadIdx.x & 31;
  int ntN  = N >> 6;
  int tM   = (wave / ntN) << 4;
  int tN   = (wave % ntN) << 6;
  if (tM >= M) return;
  int hl = lane >> 4, m16 = lane & 15;
  v8f c0 = {}, c1 = {}, c2 = {}, c3 = {};
  const u16* arow = A + (tM + m16) * K + hl * 8;      // 32-bit offsets
  const u16* brow = B + (hl * 8) * N + tN + m16;
  v16bf a  = load_frag_a(arow);
  v16bf b0 = load_frag_b(brow, N);
  v16bf b1 = load_frag_b(brow + 16, N);
  v16bf b2 = load_frag_b(brow + 32, N);
  v16bf b3 = load_frag_b(brow + 48, N);
  for (int kk = 32; kk < K; kk += 32) {
    const u16* bn = brow + kk * N;
    v16bf an = load_frag_a(arow + kk);
    v16bf n0 = load_frag_b(bn, N);
    v16bf n1 = load_frag_b(bn + 16, N);
    v16bf n2 = load_frag_b(bn + 32, N);
    v16bf n3 = load_frag_b(bn + 48, N);
    c0 = WMMA_BF16(a, b0, c0);
    c1 = WMMA_BF16(a, b1, c1);
    c2 = WMMA_BF16(a, b2, c2);
    c3 = WMMA_BF16(a, b3, c3);
    a = an; b0 = n0; b1 = n1; b2 = n2; b3 = n3;
  }
  c0 = WMMA_BF16(a, b0, c0);
  c1 = WMMA_BF16(a, b1, c1);
  c2 = WMMA_BF16(a, b2, c2);
  c3 = WMMA_BF16(a, b3, c3);
#pragma unroll
  for (int s = 0; s < 4; ++s) {
    v8f c = (s == 0) ? c0 : (s == 1) ? c1 : (s == 2) ? c2 : c3;
#pragma unroll
    for (int r = 0; r < 8; ++r) {
      int row = tM + hl * 8 + r;
      int col = tN + s * 16 + m16;
      float v = c[r];
      if (bias) v += bias[col];
      if (act == 1)      v = fmaxf(v, 0.f);
      else if (act == 2) v = (v > 0.f) ? v : 0.01f * v;
      C[row * N + col] = v;
    }
  }
}

// column sums of h (for mean over axis=0)
__global__ void k_colsum(const float* __restrict__ H, float* __restrict__ csum) {
  int d  = threadIdx.x;          // 512 threads
  int r0 = blockIdx.x * 64;
  float acc = 0.f;
  for (int i = 0; i < 64; ++i) acc += H[(r0 + i) * DD + d];
  atomicAdd(&csum[d], acc);
}

// h = (h + mean)*0.5 ; also emit bf16 copy
__global__ void k_hpost(float* __restrict__ H, const float* __restrict__ csum,
                        u16* __restrict__ Hb) {
  int i = blockIdx.x * blockDim.x + threadIdx.x;
  int d = i & (DD - 1);
  float v = (H[i] + csum[d] * (1.0f / NN)) * 0.5f;
  H[i] = v; Hb[i] = f2bf(v);
}

__global__ void k_scaleconv(const float* __restrict__ S, u16* __restrict__ Dst,
                            float sc, int n) {
  int i = blockIdx.x * blockDim.x + threadIdx.x;
  if (i < n) Dst[i] = f2bf(S[i] * sc);
}

// e_t [N,D] -> e_tT bf16 [D,N] (coalesced writes)
__global__ void k_transconv(const float* __restrict__ ET, u16* __restrict__ ETT) {
  int o  = blockIdx.x * blockDim.x + threadIdx.x;   // over D*N
  int dd = o >> 13;            // / 8192
  int nn = o & (NN - 1);
  ETT[o] = f2bf(ET[nn * DD + dd]);
}

// Fused logits GEMM + row-wise top-6.
//  * Block owns 32 rows; the 32x512 bf16 A-block is async-copied to LDS once
//    (global_load_async_to_lds_b128 + s_wait_asynccnt) and re-read as
//    ds_load_b128 fragments -> each of 256 blocks streams ETT once (2 GB L2
//    traffic total, ~balanced with the WMMA compute).
//  * 8 waves per iteration compute a 32x128 stripe: each wave reuses its B
//    fragment for 2 row-tiles (2 wmma accumulators); B pipelined one K-step.
//  * 256 scanner threads (32 rows x 8 segments) keep persistent top-6 in
//    registers; one 48-candidate merge per row at the end.
__global__ __launch_bounds__(256) void k_topk(
    const u16* __restrict__ EHS, const u16* __restrict__ ETT,
    float* __restrict__ TW, int* __restrict__ TI)
{
  __shared__ u16   As[32 * 512];          // 32 KB A-block
  __shared__ float tile[32 * 132];        // 16.5 KB logits stripe
  __shared__ float pv[32 * 8 * KTOP];
  __shared__ int   pi[32 * 8 * KTOP];
  int w = threadIdx.x >> 5, lane = threadIdx.x & 31;
  int hl = lane >> 4, m16 = lane & 15;
  int rowBase = blockIdx.x << 5;

  // ---- async copy A-block (contiguous 32 KB) into LDS ----
  {
    const void* gsrc = (const void*)(EHS + rowBase * DD);
    unsigned lbase = (unsigned)(unsigned long long)(void*)As;
    unsigned toff  = threadIdx.x * 16;
#pragma unroll
    for (int i = 0; i < 8; ++i)
      async_g2l_b128(lbase + toff + i * 4096, toff + i * 4096, gsrc);
    wait_async0();
  }
  __syncthreads();

  // persistent per-scanner top-6 (all 256 threads scan)
  float tv[KTOP]; int tix[KTOP];
#pragma unroll
  for (int k = 0; k < KTOP; ++k) { tv[k] = -INFINITY; tix[k] = 0; }
  int srow = threadIdx.x & 31;
  int sseg = threadIdx.x >> 5;

  const u16* a0p = (const u16*)&As[m16 * DD + hl * 8];          // row-tile 0
  const u16* a1p = (const u16*)&As[(16 + m16) * DD + hl * 8];   // row-tile 1

  for (int cg = 0; cg < NN; cg += 128) {
    const u16* bcol = ETT + (hl * 8) * NN + (cg + w * 16) + m16;
    if (cg + 128 < NN)
      __builtin_prefetch(bcol + 128, 0, 3);       // next stripe -> global_prefetch
    v8f c0 = {}, c1 = {};
    v16bf bcur = load_frag_b(bcol, NN);
#pragma unroll
    for (int t = 0; t < 16; ++t) {
      v16bf bnxt;
      if (t < 15) bnxt = load_frag_b(bcol + (t + 1) * 32 * NN, NN);
      v16bf a0 = load_frag_a(a0p + t * 32);   // ds_load_b128 x2
      v16bf a1 = load_frag_a(a1p + t * 32);
      c0 = WMMA_BF16(a0, bcur, c0);
      c1 = WMMA_BF16(a1, bcur, c1);
      bcur = bnxt;
    }
#pragma unroll
    for (int r = 0; r < 8; ++r) {
      tile[(hl * 8 + r) * 132 + w * 16 + m16]        = c0[r];
      tile[(16 + hl * 8 + r) * 132 + w * 16 + m16]   = c1[r];
    }
    __syncthreads();
    {
      int cbase = sseg * 16;
#pragma unroll
      for (int cc = 0; cc < 16; ++cc) {
        float v = tile[srow * 132 + cbase + cc];
        topk_insert(v, cg + cbase + cc, tv, tix);
      }
    }
    __syncthreads();
  }
  // dump partial lists, then one merge per row
#pragma unroll
  for (int k = 0; k < KTOP; ++k) {
    pv[(srow * 8 + sseg) * KTOP + k] = tv[k];
    pi[(srow * 8 + sseg) * KTOP + k] = tix[k];
  }
  __syncthreads();
  if (threadIdx.x < 32) {
    int r = threadIdx.x;
    float fv[KTOP]; int fi[KTOP];
#pragma unroll
    for (int k = 0; k < KTOP; ++k) { fv[k] = -INFINITY; fi[k] = 0; }
    for (int s = 0; s < 8; ++s)
#pragma unroll
      for (int k = 0; k < KTOP; ++k)
        topk_insert(pv[(r * 8 + s) * KTOP + k], pi[(r * 8 + s) * KTOP + k], fv, fi);
    int row = rowBase + r;
#pragma unroll
    for (int k = 0; k < KTOP; ++k) {
      TW[row * KTOP + k] = fv[k];
      TI[row * KTOP + k] = fi[k];
    }
  }
}

// One wave per row: softmax(tw), gate/tanh, ka, kp, e_Nh; emits bf16
// (e_h+e_Nh) and (e_h*e_Nh) for the lin1/lin2 GEMMs.
__global__ __launch_bounds__(256) void k_neighbor(
    const float* __restrict__ EH, const float* __restrict__ ET,
    const float* __restrict__ TW, const int* __restrict__ TI,
    u16* __restrict__ SB, u16* __restrict__ BB)
{
  int wave = (blockIdx.x * 256 + (int)threadIdx.x) >> 5;
  int lane = threadIdx.x & 31;
  if (wave >= NN) return;
  int n = wave;
  float p[KTOP]; int idx[KTOP];
  float mx = -INFINITY;
#pragma unroll
  for (int k = 0; k < KTOP; ++k) {
    p[k] = TW[n * KTOP + k]; idx[k] = TI[n * KTOP + k];
    mx = fmaxf(mx, p[k]);
  }
  float ps = 0.f;
#pragma unroll
  for (int k = 0; k < KTOP; ++k) { p[k] = expf(p[k] - mx); ps += p[k]; }
#pragma unroll
  for (int k = 0; k < KTOP; ++k) p[k] /= ps;

  float eh[16];
#pragma unroll
  for (int j = 0; j < 16; ++j) eh[j] = EH[n * DD + lane + 32 * j];

  float ka[KTOP];
#pragma unroll
  for (int k = 0; k < KTOP; ++k) {
    const float* nbp = ET + idx[k] * DD;
    float acc = 0.f;
#pragma unroll
    for (int j = 0; j < 16; ++j) {
      float nb  = nbp[lane + 32 * j];
      float ehr = p[k] * nb + (1.f - p[k]) * eh[j];
      acc += nb * tanhf(eh[j] + ehr);
    }
#pragma unroll
    for (int off = 16; off > 0; off >>= 1) acc += __shfl_xor(acc, off, 32);
    ka[k] = acc;
  }
  mx = -INFINITY;
#pragma unroll
  for (int k = 0; k < KTOP; ++k) mx = fmaxf(mx, ka[k]);
  float ks = 0.f;
#pragma unroll
  for (int k = 0; k < KTOP; ++k) { ka[k] = expf(ka[k] - mx); ks += ka[k]; }
#pragma unroll
  for (int k = 0; k < KTOP; ++k) ka[k] /= ks;

  float enh[16];
#pragma unroll
  for (int j = 0; j < 16; ++j) enh[j] = 0.f;
#pragma unroll
  for (int k = 0; k < KTOP; ++k) {
    const float* nbp = ET + idx[k] * DD;
#pragma unroll
    for (int j = 0; j < 16; ++j) enh[j] += ka[k] * nbp[lane + 32 * j];
  }
#pragma unroll
  for (int j = 0; j < 16; ++j) {
    int o = n * DD + lane + 32 * j;
    SB[o] = f2bf(eh[j] + enh[j]);
    BB[o] = f2bf(eh[j] * enh[j]);
  }
}

__global__ void k_addconv(const float* __restrict__ T1, const float* __restrict__ T2,
                          float* __restrict__ EMB, u16* __restrict__ EMBB, int n) {
  int i = blockIdx.x * blockDim.x + threadIdx.x;
  if (i < n) { float v = T1[i] + T2[i]; EMB[i] = v; EMBB[i] = f2bf(v); }
}

// att[n] = z[n,:] . att2_w + att2_b  (one wave per row)
__global__ __launch_bounds__(256) void k_attmv(
    const float* __restrict__ Z, const float* __restrict__ w2,
    const float* __restrict__ b2, float* __restrict__ ATT)
{
  int wave = (blockIdx.x * 256 + (int)threadIdx.x) >> 5;
  int lane = threadIdx.x & 31;
  if (wave >= NN) return;
  float acc = 0.f;
#pragma unroll
  for (int j = 0; j < 8; ++j) {
    int d = lane + 32 * j;
    acc += Z[wave * DH + d] * w2[d];
  }
#pragma unroll
  for (int off = 16; off > 0; off >>= 1) acc += __shfl_xor(acc, off, 32);
  if (lane == 0) ATT[wave] = acc + b2[0];
}

// softmax over all N rows (single block)
__global__ void k_softmax_n(const float* __restrict__ ATT, float* __restrict__ A) {
  __shared__ float red[1024];
  int t = threadIdx.x;
  float m = -INFINITY;
  for (int i = t; i < NN; i += 1024) m = fmaxf(m, ATT[i]);
  red[t] = m; __syncthreads();
  for (int s = 512; s > 0; s >>= 1) { if (t < s) red[t] = fmaxf(red[t], red[t + s]); __syncthreads(); }
  float M = red[0]; __syncthreads();
  float sum = 0.f;
  for (int i = t; i < NN; i += 1024) sum += expf(ATT[i] - M);
  red[t] = sum; __syncthreads();
  for (int s = 512; s > 0; s >>= 1) { if (t < s) red[t] += red[t + s]; __syncthreads(); }
  float S = red[0];
  for (int i = t; i < NN; i += 1024) A[i] = expf(ATT[i] - M) / S;
}

// pooled[d] = sum_n a[n]*emb[n,d]   (single block of 512)
__global__ void k_pooled(const float* __restrict__ Aw, const float* __restrict__ EMB,
                         float* __restrict__ P) {
  int d = threadIdx.x;
  float acc = 0.f;
  for (int nn = 0; nn < NN; ++nn) acc += Aw[nn] * EMB[nn * DD + d];
  P[d] = acc;
}

// LayerNorm + head + softmax + argmax -> d_out[5]
__global__ void k_final(const float* __restrict__ P, const float* __restrict__ g,
                        const float* __restrict__ b, const float* __restrict__ fw,
                        const float* __restrict__ fb, float* __restrict__ OUT)
{
  __shared__ float red[512];
  int d = threadIdx.x;
  float v = P[d];
  red[d] = v; __syncthreads();
  for (int s = 256; s > 0; s >>= 1) { if (d < s) red[d] += red[d + s]; __syncthreads(); }
  float mu = red[0] * (1.f / DD); __syncthreads();
  float c = v - mu;
  red[d] = c * c; __syncthreads();
  for (int s = 256; s > 0; s >>= 1) { if (d < s) red[d] += red[d + s]; __syncthreads(); }
  float var = red[0] * (1.f / DD); __syncthreads();
  float ln = c * rsqrtf(var + 1e-5f) * g[d] + b[d];
  red[d] = ln * fw[d * 2]; __syncthreads();
  for (int s = 256; s > 0; s >>= 1) { if (d < s) red[d] += red[d + s]; __syncthreads(); }
  float o0 = red[0]; __syncthreads();
  red[d] = ln * fw[d * 2 + 1]; __syncthreads();
  for (int s = 256; s > 0; s >>= 1) { if (d < s) red[d] += red[d + s]; __syncthreads(); }
  float o1 = red[0];
  if (d == 0) {
    o0 += fb[0]; o1 += fb[1];
    float m = fmaxf(o0, o1);
    float e0 = expf(o0 - m), e1 = expf(o1 - m), s = e0 + e1;
    OUT[0] = o0; OUT[1] = o1;
    OUT[2] = e0 / s; OUT[3] = e1 / s;
    OUT[4] = (o1 > o0) ? 1.f : 0.f;
  }
}

// ----------------------------------------------------------------- host ----
extern "C" void kernel_launch(void* const* d_in, const int* in_sizes, int n_in,
                              void* d_out, int out_size, void* d_ws, size_t ws_size,
                              hipStream_t stream)
{
  const float* x     = (const float*)d_in[0];
  const float* fc1_w = (const float*)d_in[1];
  const float* fc1_b = (const float*)d_in[2];
  const float* wh_w  = (const float*)d_in[3];
  const float* wh_b  = (const float*)d_in[4];
  const float* wt_w  = (const float*)d_in[5];
  const float* wt_b  = (const float*)d_in[6];
  const float* l1_w  = (const float*)d_in[7];
  const float* l1_b  = (const float*)d_in[8];
  const float* l2_w  = (const float*)d_in[9];
  const float* l2_b  = (const float*)d_in[10];
  const float* a1_w  = (const float*)d_in[11];
  const float* a1_b  = (const float*)d_in[12];
  const float* a2_w  = (const float*)d_in[13];
  const float* a2_b  = (const float*)d_in[14];
  const float* ln_g  = (const float*)d_in[15];
  const float* ln_b  = (const float*)d_in[16];
  const float* fc_w  = (const float*)d_in[17];
  const float* fc_b  = (const float*)d_in[18];
  (void)in_sizes; (void)n_in; (void)out_size; (void)ws_size;

  char* base = (char*)d_ws;
  size_t off = 0;
  auto alloc = [&](size_t bytes) -> void* {
    off = (off + 255) & ~(size_t)255;
    void* p = base + off;
    off += bytes;
    return p;
  };
  u16*   xb     = (u16*)  alloc((size_t)NN * DIN * 2);
  u16*   w1b    = (u16*)  alloc((size_t)DIN * DD * 2);
  u16*   whb    = (u16*)  alloc((size_t)DD * DD * 2);
  u16*   wtb    = (u16*)  alloc((size_t)DD * DD * 2);
  u16*   l1b    = (u16*)  alloc((size_t)DD * DD * 2);
  u16*   l2b    = (u16*)  alloc((size_t)DD * DD * 2);
  u16*   a1bw   = (u16*)  alloc((size_t)DD * DH * 2);
  float* h      = (float*)alloc((size_t)NN * DD * 4);
  u16*   hb     = (u16*)  alloc((size_t)NN * DD * 2);
  float* csum   = (float*)alloc((size_t)DD * 4);
  float* e_h    = (float*)alloc((size_t)NN * DD * 4);
  float* e_t    = (float*)alloc((size_t)NN * DD * 4);
  u16*   ehs    = (u16*)  alloc((size_t)NN * DD * 2);
  u16*   ett    = (u16*)  alloc((size_t)DD * NN * 2);
  float* tw     = (float*)alloc((size_t)NN * KTOP * 4);
  int*   ti     = (int*)  alloc((size_t)NN * KTOP * 4);
  u16*   sb     = (u16*)  alloc((size_t)NN * DD * 2);
  u16*   bb     = (u16*)  alloc((size_t)NN * DD * 2);
  float* t1     = (float*)alloc((size_t)NN * DD * 4);
  float* t2     = (float*)alloc((size_t)NN * DD * 4);
  float* emb    = (float*)alloc((size_t)NN * DD * 4);
  u16*   embb   = (u16*)  alloc((size_t)NN * DD * 2);
  float* z      = (float*)alloc((size_t)NN * DH * 4);
  float* att    = (float*)alloc((size_t)NN * 4);
  float* aw     = (float*)alloc((size_t)NN * 4);
  float* pooled = (float*)alloc((size_t)DD * 4);

  auto cvt = [&](const float* s, u16* d, int n) {
    k_f32_to_bf16<<<(n + 255) / 256, 256, 0, stream>>>(s, d, n);
  };
  cvt(x,     xb,   NN * DIN);
  cvt(fc1_w, w1b,  DIN * DD);
  cvt(wh_w,  whb,  DD * DD);
  cvt(wt_w,  wtb,  DD * DD);
  cvt(l1_w,  l1b,  DD * DD);
  cvt(l2_w,  l2b,  DD * DD);
  cvt(a1_w,  a1bw, DD * DH);

  auto gemm = [&](const u16* A, const u16* B, const float* bias, float* C,
                  int M, int Nc, int K, int act) {
    int waves = (M >> 4) * (Nc >> 6);
    k_gemm_bf16<<<waves / 8, 256, 0, stream>>>(A, B, bias, C, M, Nc, K, act);
  };

  // fc1 + relu
  gemm(xb, w1b, fc1_b, h, NN, DD, DIN, 1);
  // h = (h + mean)*0.5
  hipMemsetAsync(csum, 0, DD * 4, stream);
  k_colsum<<<NN / 64, DD, 0, stream>>>(h, csum);
  k_hpost<<<(NN * DD) / 256, 256, 0, stream>>>(h, csum, hb);
  // e_h, e_t
  gemm(hb, whb, wh_b, e_h, NN, DD, DD, 0);
  gemm(hb, wtb, wt_b, e_t, NN, DD, DD, 0);
  // logits + top-k (fused)
  k_scaleconv<<<(NN * DD) / 256, 256, 0, stream>>>(e_h, ehs, SCALEF, NN * DD);
  k_transconv<<<(NN * DD) / 256, 256, 0, stream>>>(e_t, ett);
  k_topk<<<NN / 32, 256, 0, stream>>>(ehs, ett, tw, ti);
  // neighbor aggregation
  k_neighbor<<<NN / 8, 256, 0, stream>>>(e_h, e_t, tw, ti, sb, bb);
  // lin1/lin2 + leaky, emb = sum
  gemm(sb, l1b, l1_b, t1, NN, DD, DD, 2);
  gemm(bb, l2b, l2_b, t2, NN, DD, DD, 2);
  k_addconv<<<(NN * DD) / 256, 256, 0, stream>>>(t1, t2, emb, embb, NN * DD);
  // readout
  gemm(embb, a1bw, a1_b, z, NN, DH, DD, 2);
  k_attmv<<<NN / 8, 256, 0, stream>>>(z, a2_w, a2_b, att);
  k_softmax_n<<<1, 1024, 0, stream>>>(att, aw);
  k_pooled<<<1, DD, 0, stream>>>(aw, emb, pooled);
  k_final<<<1, DD, 0, stream>>>(pooled, ln_g, ln_b, fc_w, fc_b, (float*)d_out);
}
